// FeatureMatcher_25993142075777
// MI455X (gfx1250) — matrix operand order repaired and checked by reading, not compiled, not run on hardware
//
#include <hip/hip_runtime.h>
#include <hip/hip_bf16.h>
#include <math.h>

// ---------------------------------------------------------------------------
// FeatureMatcher for MI455X (gfx1250, wave32, WMMA)
//   phase 1: p1 = feat1 @ W1[:512], p2 = feat2 @ W1[512:]        (f32, tiny)
//   phase 2: per 16x16 (i,j) tile:
//     h1 = relu(p1+p2+b1) -> f16 LDS (double buffered, built one tile ahead)
//     h2pre = h1(16x256) @ W2 via 8x v_wmma_f32_16x16x32_f16 (B strip in VGPRs)
//     h2 = relu(h2pre+b2) -> f16 LDS
//     sim = h2(16x128) @ W3 via 4x WMMA (W3 in column 0 of B), wave 0 only
//     sigmoid(+b3) -> out
// ---------------------------------------------------------------------------

typedef __attribute__((ext_vector_type(16))) _Float16 v16h;
typedef __attribute__((ext_vector_type(8)))  _Float16 v8h;
typedef __attribute__((ext_vector_type(8)))  float    v8f;
typedef __attribute__((ext_vector_type(4)))  float    v4f;

#define FD      512
#define H       256
#define H2      128
#define NROWS   1024
#define P_STRIDE  260   // f32 row pitch for staged p1/p2 (bank spread)
#define H1_STRIDE 264   // f16 row pitch for h1 (528B: 16B aligned, 4-bank lane stride)
#define H2_STRIDE 136   // f16 row pitch for h2 (272B: 16B aligned, 4-bank lane stride)

// ---------------- phase 1: feature projections ----------------
__global__ __launch_bounds__(256)
void fm_proj_kernel(const float* __restrict__ feat1,
                    const float* __restrict__ feat2,
                    const float* __restrict__ W1,
                    float* __restrict__ p1,
                    float* __restrict__ p2) {
  const int n     = threadIdx.x;
  const int r0    = blockIdx.x * 8;
  const int which = blockIdx.y;
  const float* feat = which ? feat2 : feat1;
  const float* w    = W1 + (size_t)which * FD * H + n;
  float acc[8];
#pragma unroll
  for (int r = 0; r < 8; ++r) acc[r] = 0.f;
  for (int k = 0; k < FD; ++k) {
    const float wv = w[(size_t)k * H];
#pragma unroll
    for (int r = 0; r < 8; ++r)
      acc[r] = fmaf(feat[(size_t)(r0 + r) * FD + k], wv, acc[r]);
  }
  float* pout = which ? p2 : p1;
#pragma unroll
  for (int r = 0; r < 8; ++r) pout[(size_t)(r0 + r) * H + n] = acc[r];
}

// ---------------- phase 2: pairwise MLP (WMMA) ----------------
__global__ __launch_bounds__(256)
void fm_main_kernel(const float* __restrict__ p1,
                    const float* __restrict__ p2,
                    const float* __restrict__ b1,
                    const float* __restrict__ W2,
                    const float* __restrict__ b2,
                    const float* __restrict__ W3,
                    const float* __restrict__ b3,
                    float* __restrict__ out) {
  __shared__ float s_p1[16 * P_STRIDE];
  __shared__ float s_p2[16 * P_STRIDE];
  __shared__ float s_b1[H];
  __shared__ __align__(16) _Float16 s_h1[2][16 * H1_STRIDE];  // double buffered
  __shared__ __align__(16) _Float16 s_h2[16 * H2_STRIDE];
  __shared__ float s_sim[256];

  const int tid  = threadIdx.x;
  const int lane = tid & 31;
  const int wave = tid >> 5;            // 8 waves -> 8 N-strips of 16 cols
  const int i0   = blockIdx.x * 16;
  const int j0   = blockIdx.y * 16;

  // ---- stage p1/p2 rows, b1 ----
#pragma unroll
  for (int u = 0; u < 16; ++u) {
    s_p1[u * P_STRIDE + tid] = p1[(size_t)(i0 + u) * H + tid];
    s_p2[u * P_STRIDE + tid] = p2[(size_t)(j0 + u) * H + tid];
  }
  s_b1[tid] = b1[tid];

  // ---- per-lane constants + resident B fragments ----
  // CDNA5 16-bit B layout (KxN=32x16): lane holds N=lane&15;
  // lanes 0-15 hold K 0..7,16..23 of the chunk, lanes 16-31 hold K 8..15,24..31.
  const int   ncol  = wave * 16 + (lane & 15);
  const float b2v   = b2[ncol];
  const int   kband = (lane >> 4) << 3;    // 0 or 8 within a 32-K chunk
  const bool  isc0  = (lane & 15) == 0;

  v16h bfrag[8];                           // W2 strip for this wave (64 VGPRs)
#pragma unroll
  for (int c = 0; c < 8; ++c) {
    const float* col = W2 + (size_t)(c * 32 + kband) * H2 + ncol;
    v16h b;
#pragma unroll
    for (int h = 0; h < 16; ++h) b[h] = (_Float16)col[(size_t)(h + (h & 8)) * H2];
    bfrag[c] = b;
  }
  v16h w3frag[4];                          // W3 in column 0, zeros elsewhere
#pragma unroll
  for (int c = 0; c < 4; ++c) {
    v16h b;
#pragma unroll
    for (int h = 0; h < 16; ++h)
      b[h] = isc0 ? (_Float16)W3[c * 32 + kband + h + (h & 8)] : (_Float16)0.0f;
    w3frag[c] = b;
  }

  const int jl = tid & 15;                // pair (j) row when building h1
  const int ks = tid >> 4;                // 16-element k segment when building h1
  const int m  = lane & 15;               // A-matrix row for this lane
  const int koffbase = (lane >> 4) << 3;  // A layout: lanes>=16 start at K+8

  auto build_h1 = [&](int t, _Float16* dst) {
    v8h hv0, hv1;
#pragma unroll
    for (int u = 0; u < 8; ++u) {
      const int k = ks * 16 + u;
      float v = s_p1[t * P_STRIDE + k] + s_p2[jl * P_STRIDE + k] + s_b1[k];
      hv0[u] = (_Float16)fmaxf(v, 0.f);
    }
#pragma unroll
    for (int u = 0; u < 8; ++u) {
      const int k = ks * 16 + 8 + u;
      float v = s_p1[t * P_STRIDE + k] + s_p2[jl * P_STRIDE + k] + s_b1[k];
      hv1[u] = (_Float16)fmaxf(v, 0.f);
    }
    *(v8h*)&dst[jl * H1_STRIDE + ks * 16]     = hv0;
    *(v8h*)&dst[jl * H1_STRIDE + ks * 16 + 8] = hv1;
  };

  __syncthreads();                        // staged p1/p2/b1 visible
  build_h1(0, &s_h1[0][0]);

  for (int t = 0; t < 16; ++t) {
    __syncthreads();                      // B1: h1[t] ready, s_h2 free

    // ---- main GEMM: 16x16 h2pre = h1(16x256) @ W2strip(256x16) ----
    // CDNA5 16-bit A layout (MxK=16x32): lane m=lane&15;
    //   lanes 0-15: halves 0-7 -> K 0..7,  8-15 -> K 16..23
    //   lanes 16-31: halves 0-7 -> K 8..15, 8-15 -> K 24..31
    const _Float16* hrow = &s_h1[t & 1][m * H1_STRIDE];
    v8f acc = {};
    v8h lo = *(const v8h*)(hrow + koffbase);
    v8h hi = *(const v8h*)(hrow + koffbase + 16);
    v16h a_cur = __builtin_shufflevector(lo, hi,
                   0,1,2,3,4,5,6,7,8,9,10,11,12,13,14,15);
#pragma unroll
    for (int c = 0; c < 8; ++c) {
      v16h a_next = a_cur;
      if (c < 7) {                        // prefetch next chunk before WMMA
        const int koff = (c + 1) * 32 + koffbase;
        v8h nlo = *(const v8h*)(hrow + koff);
        v8h nhi = *(const v8h*)(hrow + koff + 16);
        a_next = __builtin_shufflevector(nlo, nhi,
                   0,1,2,3,4,5,6,7,8,9,10,11,12,13,14,15);
      }
      acc = __builtin_amdgcn_wmma_f32_16x16x32_f16(
                false, a_cur, false, bfrag[c], (short)0, acc, false, false);
      a_cur = a_next;
    }

    // ---- h2 = relu(h2pre + b2) -> f16 LDS (C layout: M = r + koffbase) ----
#pragma unroll
    for (int r = 0; r < 8; ++r) {
      const float h2v = fmaxf(acc[r] + b2v, 0.f);
      s_h2[(r + koffbase) * H2_STRIDE + ncol] = (_Float16)h2v;
    }

    // ---- overlap: build next tile's h1 into the other buffer ----
    if (t < 15) build_h1(t + 1, &s_h1[(t + 1) & 1][0]);

    __syncthreads();                      // B2: s_h2 (and h1[t+1]) ready

    // ---- W3 matvec as WMMA, wave 0 only: sim[16] = h2(16x128) @ w3col ----
    if (wave == 0) {
      v8f acc2 = {};
      const _Float16* h2row = &s_h2[m * H2_STRIDE];
#pragma unroll
      for (int c = 0; c < 4; ++c) {
        const int koff = c * 32 + koffbase;
        v8h alo = *(const v8h*)(h2row + koff);
        v8h ahi = *(const v8h*)(h2row + koff + 16);
        v16h a = __builtin_shufflevector(alo, ahi,
                   0,1,2,3,4,5,6,7,8,9,10,11,12,13,14,15);
        acc2 = __builtin_amdgcn_wmma_f32_16x16x32_f16(
                   false, a, false, w3frag[c], (short)0, acc2, false, false);
      }
      // column 0 lives in lanes 0 (M=r) and 16 (M=r+8): 8 contiguous floats
      if (isc0) {
        v4f lo4 = __builtin_shufflevector(acc2, acc2, 0, 1, 2, 3);
        v4f hi4 = __builtin_shufflevector(acc2, acc2, 4, 5, 6, 7);
        *(v4f*)&s_sim[t * 16 + koffbase]     = lo4;
        *(v4f*)&s_sim[t * 16 + koffbase + 4] = hi4;
      }
    }
  }
  __syncthreads();

  // ---- sigmoid(+b3) and store: pair tid -> (i0 + tid>>4, j0 + tid&15) ----
  const float x   = s_sim[tid] + b3[0];
  const float sim = 1.f / (1.f + __expf(-x));
  out[(size_t)(i0 + (tid >> 4)) * NROWS + (j0 + (tid & 15))] = sim;
}

// ---------------------------------------------------------------------------
extern "C" void kernel_launch(void* const* d_in, const int* in_sizes, int n_in,
                              void* d_out, int out_size, void* d_ws, size_t ws_size,
                              hipStream_t stream) {
  const float* feat1 = (const float*)d_in[0];   // [1024,512]
  const float* feat2 = (const float*)d_in[1];   // [1024,512]
  const float* W1    = (const float*)d_in[2];   // [1024,256]
  const float* b1    = (const float*)d_in[3];   // [256]
  const float* W2    = (const float*)d_in[4];   // [256,128]
  const float* b2    = (const float*)d_in[5];   // [128]
  const float* W3    = (const float*)d_in[6];   // [128]
  const float* b3    = (const float*)d_in[7];   // [1]
  float* out = (float*)d_out;                   // [1024,1024] f32

  float* p1 = (float*)d_ws;                     // [1024,256] f32
  float* p2 = p1 + (size_t)NROWS * H;           // [1024,256] f32

  fm_proj_kernel<<<dim3(NROWS / 8, 2), 256, 0, stream>>>(feat1, feat2, W1, p1, p2);
  fm_main_kernel<<<dim3(NROWS / 16, NROWS / 16), 256, 0, stream>>>(
      p1, p2, b1, W2, b2, W3, b3, out);
}
